// BERT_CRF_NER_798863917465
// MI455X (gfx1250) — matrix operand order repaired and checked by reading, not compiled, not run on hardware
//
#include <hip/hip_runtime.h>
#include <hip/hip_bf16.h>
#include <math.h>

// ---------------- problem constants ----------------
#define V_   30522
#define TMAX 512
#define H_   768
#define L_   11
#define F_   3072
#define NHEAD_ 12
#define DH_  64
#define NL_  11
#define START_ID_ 9
#define B_   16
#define T_   256
#define NEG_ (-10000.0f)

typedef __attribute__((ext_vector_type(16))) _Float16 v16h;
typedef __attribute__((ext_vector_type(8)))  float    v8f;

// ---------------- helpers ----------------
__device__ inline float block_sum256(float v, float* red) {
    for (int o = 16; o > 0; o >>= 1) v += __shfl_down(v, o, 32);
    int lane = threadIdx.x & 31, w = threadIdx.x >> 5;
    if (lane == 0) red[w] = v;
    __syncthreads();
    float t = (lane < 8) ? red[lane] : 0.f;
    if (w == 0) {
        for (int o = 4; o > 0; o >>= 1) t += __shfl_down(t, o, 32);
        if (lane == 0) red[0] = t;
    }
    __syncthreads();
    float r = red[0];
    __syncthreads();
    return r;
}

__device__ inline float block_max256(float v, float* red) {
    for (int o = 16; o > 0; o >>= 1) v = fmaxf(v, __shfl_down(v, o, 32));
    int lane = threadIdx.x & 31, w = threadIdx.x >> 5;
    if (lane == 0) red[w] = v;
    __syncthreads();
    float t = (lane < 8) ? red[lane] : -3.4e38f;
    if (w == 0) {
        for (int o = 4; o > 0; o >>= 1) t = fmaxf(t, __shfl_down(t, o, 32));
        if (lane == 0) red[0] = t;
    }
    __syncthreads();
    float r = red[0];
    __syncthreads();
    return r;
}

// ---------------- f32 -> f16 conversion ----------------
__global__ void cvt_f32_f16_kernel(const float* __restrict__ src,
                                   _Float16* __restrict__ dst, long n) {
    long i = (long)blockIdx.x * blockDim.x + threadIdx.x;
    long stride = (long)gridDim.x * blockDim.x;
    for (; i < n; i += stride) dst[i] = (_Float16)src[i];
}

// ---------------- embedding gather ----------------
__global__ __launch_bounds__(256)
void embed_kernel(const int* __restrict__ ids, const int* __restrict__ segs,
                  const float* __restrict__ we, const float* __restrict__ pe,
                  const float* __restrict__ te, float* __restrict__ out) {
    int tok = blockIdx.x;
    int t = tok % T_;
    int id = ids[tok];
    int sg = segs[tok];
    for (int i = threadIdx.x; i < H_; i += blockDim.x)
        out[(size_t)tok * H_ + i] =
            we[(size_t)id * H_ + i] + pe[(size_t)t * H_ + i] + te[(size_t)sg * H_ + i];
}

// ---------------- residual + layernorm (two-pass like reference) ----------------
__global__ __launch_bounds__(256)
void resid_ln_kernel(const float* __restrict__ resid, const float* __restrict__ y,
                     const float* __restrict__ g, const float* __restrict__ bta,
                     float* __restrict__ out32, _Float16* __restrict__ out16) {
    __shared__ float red[8];
    int row = blockIdx.x;
    float v[3];
#pragma unroll
    for (int i = 0; i < 3; ++i) {
        int col = threadIdx.x + i * 256;
        size_t idx = (size_t)row * H_ + col;
        v[i] = resid[idx] + (y ? y[idx] : 0.f);
    }
    float s = block_sum256(v[0] + v[1] + v[2], red);
    float mu = s * (1.0f / H_);
    float q = 0.f;
#pragma unroll
    for (int i = 0; i < 3; ++i) { float d = v[i] - mu; q += d * d; }
    q = block_sum256(q, red);
    float inv = rsqrtf(q * (1.0f / H_) + 1e-12f);
#pragma unroll
    for (int i = 0; i < 3; ++i) {
        int col = threadIdx.x + i * 256;
        size_t idx = (size_t)row * H_ + col;
        float o = (v[i] - mu) * inv * g[col] + bta[col];
        out32[idx] = o;
        out16[idx] = (_Float16)o;
    }
}

// ---------------- generic batched WMMA GEMM, 2-stage LDS pipeline ----------------
// C[M,N] = A[M,K] @ B[K,N] (+ bias, optional exact GELU).
// A: row-major, stride lda; staged by CDNA5 async global->LDS DMA (ASYNCcnt).
// B: element (k,n) at k*bRowStride + n*bColStride; staged transposed (Bt[n][k])
//    through VGPRs so fragments are contiguous 2x ds_load_b128 per lane.
// Double-buffered: tile k+1 DMA/loads issued before the 8 WMMAs of tile k,
// B ds_stores issued after them; one barrier per k-step.
__global__ __launch_bounds__(256)
void gemm_wmma_f16(const _Float16* __restrict__ A, long aOuter, long aInner, int lda,
                   const _Float16* __restrict__ Bp, long bOuter, long bInner,
                   int bRowStride, int bColStride,
                   float* __restrict__ C32, _Float16* __restrict__ C16,
                   long cOuter, long cInner, int ldc,
                   const float* __restrict__ bias,
                   int M, int N, int K, int innerCnt, int doGelu) {
    __shared__ __align__(16) _Float16 As[2][128][40];   // [m][k], 80 B rows
    __shared__ __align__(32) _Float16 Bt[2][128][48];   // [n][k], 96 B rows

    int tid = threadIdx.x;
    long z = blockIdx.z;
    long zo = z / innerCnt, zi = z % innerCnt;
    A  += zo * aOuter + zi * aInner;
    Bp += zo * bOuter + zi * bInner;
    long cOff = zo * cOuter + zi * cInner;

    int m0 = blockIdx.y * 128;
    int n0 = blockIdx.x * 128;
    int lane = tid & 31, w = tid >> 5;
    int wm = w >> 2, wn = w & 3;       // 2 x 4 wave grid; each wave: 64 x 32
    int half = lane >> 4, l16 = lane & 15;

    // ---- staging helpers ----
    auto stage_a = [&](int buf, int k0) {   // async DMA, 16 B per lane
        unsigned long long abase =
            (unsigned long long)(const void*)(A + (size_t)m0 * lda + k0);
#pragma unroll
        for (int i = 0; i < 2; ++i) {
            int c = tid + i * 256;
            int r = c >> 2, kc = (c & 3) * 8;
            unsigned voff = ((unsigned)r * (unsigned)lda + (unsigned)kc) * 2u;
            unsigned ldsoff = (unsigned)(unsigned long long)(uintptr_t)&As[buf][r][kc];
            asm volatile("global_load_async_to_lds_b128 %0, %1, %2"
                         :: "v"(ldsoff), "v"(voff), "s"(abase) : "memory");
        }
    };
    _Float16 bregs[16];
    auto load_b = [&](int k0) {             // global -> VGPR
        if (bColStride == 1) {
#pragma unroll
            for (int i = 0; i < 2; ++i) {
                int c = tid + i * 256;
                int r = c >> 4, nc = (c & 15) * 8;
                uint4 d = make_uint4(0u, 0u, 0u, 0u);
                if (n0 + nc + 8 <= N)
                    d = *(const uint4*)(Bp + (size_t)(k0 + r) * bRowStride + (n0 + nc));
                const _Float16* dv = (const _Float16*)&d;
#pragma unroll
                for (int j = 0; j < 8; ++j) bregs[i * 8 + j] = dv[j];
            }
        } else {
#pragma unroll
            for (int i = 0; i < 16; ++i) {
                int c = tid + i * 256;
                int n = c & 127, r = c >> 7;
                _Float16 vv = (_Float16)0.f;
                if (n0 + n < N)
                    vv = Bp[(size_t)(k0 + r) * bRowStride + (size_t)(n0 + n) * bColStride];
                bregs[i] = vv;
            }
        }
    };
    auto store_b = [&](int buf) {           // VGPR -> LDS transposed
        if (bColStride == 1) {
#pragma unroll
            for (int i = 0; i < 2; ++i) {
                int c = tid + i * 256;
                int r = c >> 4, nc = (c & 15) * 8;
#pragma unroll
                for (int j = 0; j < 8; ++j) Bt[buf][nc + j][r] = bregs[i * 8 + j];
            }
        } else {
#pragma unroll
            for (int i = 0; i < 16; ++i) {
                int c = tid + i * 256;
                int n = c & 127, r = c >> 7;
                Bt[buf][n][r] = bregs[i];
            }
        }
    };

    v8f acc[4][2];
#pragma unroll
    for (int mt = 0; mt < 4; ++mt)
#pragma unroll
        for (int nt = 0; nt < 2; ++nt)
#pragma unroll
            for (int r = 0; r < 8; ++r) acc[mt][nt][r] = 0.f;

    // ---- prologue: stage tile 0 into buffer 0 ----
    stage_a(0, 0);
    load_b(0);
    store_b(0);
    int cur = 0;

    for (int k0 = 0; k0 < K; k0 += 32) {
        asm volatile("s_wait_asynccnt 0" ::: "memory");  // this wave's A-DMA landed
        __syncthreads();   // staged tile visible; prior readers of other buffer done

        int nxt = cur ^ 1;
        bool more = (k0 + 32 < K);
        if (more) {
            stage_a(nxt, k0 + 32);   // DMA overlaps the WMMA burst below
            load_b(k0 + 32);         // loadcnt wait lands after the WMMAs
        }

        // ---- compute tile `cur`: fragments are contiguous ds_load_b128 runs ----
        v16h bf[2];
#pragma unroll
        for (int nt = 0; nt < 2; ++nt) {
            int n = wn * 32 + nt * 16 + l16;
            bf[nt] = *(const v16h*)&Bt[cur][n][16 * half];
        }
#pragma unroll
        for (int mt = 0; mt < 4; ++mt) {
            v16h af;
            int m = wm * 64 + mt * 16 + l16;
#pragma unroll
            for (int v = 0; v < 8; ++v) {
                int kb = 2 * v + 16 * half;
                af[2 * v]     = As[cur][m][kb];
                af[2 * v + 1] = As[cur][m][kb + 1];
            }
            acc[mt][0] = __builtin_amdgcn_wmma_f32_16x16x32_f16(
                false, af, false, bf[0], (short)0, acc[mt][0], false, false);
            acc[mt][1] = __builtin_amdgcn_wmma_f32_16x16x32_f16(
                false, af, false, bf[1], (short)0, acc[mt][1], false, false);
        }

        if (more) store_b(nxt);      // after compute; done before next barrier
        cur = nxt;
    }

    // ---- epilogue: bias (+ GELU), write f32 and/or f16 ----
#pragma unroll
    for (int mt = 0; mt < 4; ++mt)
#pragma unroll
        for (int nt = 0; nt < 2; ++nt) {
            int gn = n0 + wn * 32 + nt * 16 + l16;
            if (gn >= N) continue;
            float bv = bias ? bias[gn] : 0.f;
#pragma unroll
            for (int r = 0; r < 8; ++r) {
                int gm = m0 + wm * 64 + mt * 16 + r + 8 * half;
                float val = acc[mt][nt][r] + bv;
                if (doGelu) val = 0.5f * val * (1.f + erff(val * 0.70710678118654752f));
                size_t idx = (size_t)cOff + (size_t)gm * ldc + gn;
                if (C32) C32[idx] = val;
                if (C16) C16[idx] = (_Float16)val;
            }
        }
}

// ---------------- softmax over keys (row length T_) ----------------
__global__ __launch_bounds__(256)
void softmax_kernel(const float* __restrict__ S, _Float16* __restrict__ P,
                    const int* __restrict__ mask) {
    __shared__ float red[8];
    int row = blockIdx.x;                  // (b*NHEAD + h)*T + q
    int b = row / (NHEAD_ * T_);
    int k = threadIdx.x;
    float bias = (1.f - (float)mask[b * T_ + k]) * NEG_;
    float v = S[(size_t)row * T_ + k] * 0.125f + bias;
    float m = block_max256(v, red);
    float e = expf(v - m);
    float s = block_sum256(e, red);
    P[(size_t)row * T_ + k] = (_Float16)(e / s);
}

// ---------------- output projection H -> NL ----------------
__global__ __launch_bounds__(352)
void feats_kernel(const float* __restrict__ h, const float* __restrict__ outW,
                  const float* __restrict__ outB, float* __restrict__ feats) {
    __shared__ float hr[H_];
    int tok = blockIdx.x;
    for (int i = threadIdx.x; i < H_; i += blockDim.x)
        hr[i] = h[(size_t)tok * H_ + i];
    __syncthreads();
    int w = threadIdx.x >> 5, lane = threadIdx.x & 31;   // 11 waves = 11 labels
    float s = 0.f;
    for (int j = lane; j < H_; j += 32) s += hr[j] * outW[(size_t)j * NL_ + w];
    for (int o = 16; o > 0; o >>= 1) s += __shfl_down(s, o, 32);
    if (lane == 0) feats[(size_t)tok * NL_ + w] = s + outB[w];
}

// ---------------- CRF Viterbi: one wave per batch element ----------------
__global__ __launch_bounds__(512)
void viterbi_kernel(const float* __restrict__ feats, const float* __restrict__ trans,
                    int* __restrict__ psi, float* __restrict__ out) {
    __shared__ float tr[NL_ * NL_];
    int tid = threadIdx.x;
    if (tid < NL_ * NL_) tr[tid] = trans[tid];
    __syncthreads();
    int b = tid >> 5, lane = tid & 31;
    float delta = (lane == START_ID_) ? 0.f : NEG_;
    for (int t = 1; t < T_; ++t) {
        float best = -3.4e38f; int arg = 0;
#pragma unroll
        for (int i = 0; i < NL_; ++i) {
            float di = __shfl(delta, i, 32);
            float v = di + ((lane < NL_) ? tr[lane * NL_ + i] : 0.f);
            if (v > best) { best = v; arg = i; }
        }
        float nd = NEG_;
        if (lane < NL_) {
            nd = best + feats[((size_t)(b * T_ + t)) * NL_ + lane];
            psi[((size_t)(t - 1) * B_ + b) * NL_ + lane] = arg;
        }
        delta = nd;
    }
    float best = -3.4e38f; int last = 0;
#pragma unroll
    for (int i = 0; i < NL_; ++i) {
        float di = __shfl(delta, i, 32);
        if (di > best) { best = di; last = i; }
    }
    if (lane == 0) {
        out[b] = best;
        float* path = out + B_;
        int cur = last;
        path[b * T_ + (T_ - 1)] = (float)cur;
        for (int t = T_ - 2; t >= 0; --t) {
            cur = psi[((size_t)t * B_ + b) * NL_ + cur];
            path[b * T_ + t] = (float)cur;
        }
    }
}

// ---------------- host-side launcher ----------------
static void launch_gemm(hipStream_t s,
                        const _Float16* A, long aO, long aI, int lda,
                        const _Float16* Bm, long bO, long bI, int brs, int bcs,
                        float* C32, _Float16* C16, long cO, long cI, int ldc,
                        const float* bias, int M, int N, int K,
                        int batch, int innerCnt, int gelu) {
    dim3 grid((N + 127) / 128, (M + 127) / 128, batch);
    gemm_wmma_f16<<<grid, 256, 0, s>>>(A, aO, aI, lda, Bm, bO, bI, brs, bcs,
                                       C32, C16, cO, cI, ldc, bias,
                                       M, N, K, innerCnt, gelu);
}

extern "C" void kernel_launch(void* const* d_in, const int* in_sizes, int n_in,
                              void* d_out, int out_size, void* d_ws, size_t ws_size,
                              hipStream_t stream) {
    (void)in_sizes; (void)n_in; (void)out_size; (void)ws_size;
    // inputs in setup_inputs() dict order (params flattened depth-first)
    const int*   input_ids = (const int*)d_in[0];
    const int*   seg_ids   = (const int*)d_in[1];
    const int*   in_mask   = (const int*)d_in[2];
    const float* word_emb  = (const float*)d_in[3];
    const float* pos_emb   = (const float*)d_in[4];
    const float* type_emb  = (const float*)d_in[5];
    const float* emb_ln_g  = (const float*)d_in[6];
    const float* emb_ln_b  = (const float*)d_in[7];
    const float* Wq = (const float*)d_in[8];   const float* bq = (const float*)d_in[9];
    const float* Wk = (const float*)d_in[10];  const float* bk = (const float*)d_in[11];
    const float* Wv = (const float*)d_in[12];  const float* bv = (const float*)d_in[13];
    const float* Wo = (const float*)d_in[14];  const float* bo = (const float*)d_in[15];
    const float* ln1_g = (const float*)d_in[16]; const float* ln1_b = (const float*)d_in[17];
    const float* W1 = (const float*)d_in[18];  const float* b1 = (const float*)d_in[19];
    const float* W2 = (const float*)d_in[20];  const float* b2 = (const float*)d_in[21];
    const float* ln2_g = (const float*)d_in[22]; const float* ln2_b = (const float*)d_in[23];
    const float* outW = (const float*)d_in[24]; const float* outB = (const float*)d_in[25];
    const float* trans = (const float*)d_in[26];

    // workspace carve-up
    char* ws = (char*)d_ws;
    size_t off = 0;
    auto wsAlloc = [&](size_t bytes) -> char* {
        char* p = ws + off;
        off = (off + bytes + 255) & ~(size_t)255;
        return p;
    };
    const long nTok = (long)B_ * T_;
    _Float16* wq16 = (_Float16*)wsAlloc((size_t)L_ * H_ * H_ * 2);
    _Float16* wk16 = (_Float16*)wsAlloc((size_t)L_ * H_ * H_ * 2);
    _Float16* wv16 = (_Float16*)wsAlloc((size_t)L_ * H_ * H_ * 2);
    _Float16* wo16 = (_Float16*)wsAlloc((size_t)L_ * H_ * H_ * 2);
    _Float16* w116 = (_Float16*)wsAlloc((size_t)L_ * H_ * F_ * 2);
    _Float16* w216 = (_Float16*)wsAlloc((size_t)L_ * F_ * H_ * 2);
    float*    x32  = (float*)   wsAlloc((size_t)nTok * H_ * 4);
    _Float16* x16  = (_Float16*)wsAlloc((size_t)nTok * H_ * 2);
    float*    tmp32= (float*)   wsAlloc((size_t)nTok * H_ * 4);
    float*    h32  = (float*)   wsAlloc((size_t)nTok * H_ * 4);
    _Float16* h16  = (_Float16*)wsAlloc((size_t)nTok * H_ * 2);
    _Float16* q16  = (_Float16*)wsAlloc((size_t)nTok * H_ * 2);
    _Float16* k16  = (_Float16*)wsAlloc((size_t)nTok * H_ * 2);
    _Float16* v16  = (_Float16*)wsAlloc((size_t)nTok * H_ * 2);
    float*    S32  = (float*)   wsAlloc((size_t)B_ * NHEAD_ * T_ * T_ * 4);
    _Float16* P16  = (_Float16*)wsAlloc((size_t)B_ * NHEAD_ * T_ * T_ * 2);
    _Float16* c16  = (_Float16*)wsAlloc((size_t)nTok * H_ * 2);
    _Float16* g16  = (_Float16*)wsAlloc((size_t)nTok * F_ * 2);
    float*    feats= (float*)   wsAlloc((size_t)nTok * NL_ * 4);
    int*      psi  = (int*)     wsAlloc((size_t)T_ * B_ * NL_ * 4);

    // 1) weights -> f16 (once per launch; ~0.5 GB traffic, negligible vs GEMMs)
    cvt_f32_f16_kernel<<<2048, 256, 0, stream>>>(Wq, wq16, (long)L_ * H_ * H_);
    cvt_f32_f16_kernel<<<2048, 256, 0, stream>>>(Wk, wk16, (long)L_ * H_ * H_);
    cvt_f32_f16_kernel<<<2048, 256, 0, stream>>>(Wv, wv16, (long)L_ * H_ * H_);
    cvt_f32_f16_kernel<<<2048, 256, 0, stream>>>(Wo, wo16, (long)L_ * H_ * H_);
    cvt_f32_f16_kernel<<<2048, 256, 0, stream>>>(W1, w116, (long)L_ * H_ * F_);
    cvt_f32_f16_kernel<<<2048, 256, 0, stream>>>(W2, w216, (long)L_ * F_ * H_);

    // 2) embeddings + LN
    embed_kernel<<<nTok, 256, 0, stream>>>(input_ids, seg_ids, word_emb, pos_emb, type_emb, tmp32);
    resid_ln_kernel<<<nTok, 256, 0, stream>>>(tmp32, nullptr, emb_ln_g, emb_ln_b, x32, x16);

    // 3) transformer layers
    for (int l = 0; l < L_; ++l) {
        const _Float16* Wq_l = wq16 + (size_t)l * H_ * H_;
        const _Float16* Wk_l = wk16 + (size_t)l * H_ * H_;
        const _Float16* Wv_l = wv16 + (size_t)l * H_ * H_;
        const _Float16* Wo_l = wo16 + (size_t)l * H_ * H_;
        const _Float16* W1_l = w116 + (size_t)l * H_ * F_;
        const _Float16* W2_l = w216 + (size_t)l * F_ * H_;
        const float* bq_l = bq + (size_t)l * H_;
        const float* bk_l = bk + (size_t)l * H_;
        const float* bv_l = bv + (size_t)l * H_;
        const float* bo_l = bo + (size_t)l * H_;
        const float* b1_l = b1 + (size_t)l * F_;
        const float* b2_l = b2 + (size_t)l * H_;

        // QKV projections: [4096,768] @ [768,768] -> f16
        launch_gemm(stream, x16, 0, 0, H_, Wq_l, 0, 0, H_, 1,
                    nullptr, q16, 0, 0, H_, bq_l, (int)nTok, H_, H_, 1, 1, 0);
        launch_gemm(stream, x16, 0, 0, H_, Wk_l, 0, 0, H_, 1,
                    nullptr, k16, 0, 0, H_, bk_l, (int)nTok, H_, H_, 1, 1, 0);
        launch_gemm(stream, x16, 0, 0, H_, Wv_l, 0, 0, H_, 1,
                    nullptr, v16, 0, 0, H_, bv_l, (int)nTok, H_, H_, 1, 1, 0);

        // scores: per (b,h): Q[256,64] @ K^T[64,256] -> f32 (scale in softmax)
        launch_gemm(stream,
                    q16, (long)T_ * H_, DH_, H_,
                    k16, (long)T_ * H_, DH_, 1, H_,
                    S32, nullptr, (long)NHEAD_ * T_ * T_, (long)T_ * T_, T_,
                    nullptr, T_, T_, DH_, B_ * NHEAD_, NHEAD_, 0);

        softmax_kernel<<<B_ * NHEAD_ * T_, 256, 0, stream>>>(S32, P16, in_mask);

        // ctx: per (b,h): P[256,256] @ V[256,64] -> f16 into [B,T,H] head slice
        launch_gemm(stream,
                    P16, (long)NHEAD_ * T_ * T_, (long)T_ * T_, T_,
                    v16, (long)T_ * H_, DH_, H_, 1,
                    nullptr, c16, (long)T_ * H_, DH_, H_,
                    nullptr, T_, DH_, T_, B_ * NHEAD_, NHEAD_, 0);

        // attention output projection -> f32, then residual + LN1
        launch_gemm(stream, c16, 0, 0, H_, Wo_l, 0, 0, H_, 1,
                    tmp32, nullptr, 0, 0, H_, bo_l, (int)nTok, H_, H_, 1, 1, 0);
        resid_ln_kernel<<<nTok, 256, 0, stream>>>(x32, tmp32,
                                                  ln1_g + (size_t)l * H_, ln1_b + (size_t)l * H_,
                                                  h32, h16);

        // FFN: GELU fused into first GEMM epilogue
        launch_gemm(stream, h16, 0, 0, H_, W1_l, 0, 0, F_, 1,
                    nullptr, g16, 0, 0, F_, b1_l, (int)nTok, F_, H_, 1, 1, 1);
        launch_gemm(stream, g16, 0, 0, F_, W2_l, 0, 0, H_, 1,
                    tmp32, nullptr, 0, 0, H_, b2_l, (int)nTok, H_, F_, 1, 1, 0);
        resid_ln_kernel<<<nTok, 256, 0, stream>>>(h32, tmp32,
                                                  ln2_g + (size_t)l * H_, ln2_b + (size_t)l * H_,
                                                  x32, x16);
    }

    // 4) feats + Viterbi -> d_out = [score (16 floats) | path (16*256 floats)]
    feats_kernel<<<nTok, 352, 0, stream>>>(x32, outW, outB, feats);
    viterbi_kernel<<<1, 512, 0, stream>>>(feats, trans, psi, (float*)d_out);
}
